// CrossModalAttention_55757265437143
// MI455X (gfx1250) — matrix operand order repaired and checked by reading, not compiled, not run on hardware
//
#include <hip/hip_runtime.h>

#define LTOK 4096      // 16*16*16 spatial tokens
#define CIN  64
#define TD   128       // TOKEN_DIM
#define NH   4
#define HD   32        // head dim == WMMA K for f16. perfect fit.

// logit scale (1/sqrt(32)) with log2(e) folded in: softmax runs in exp2 domain.
#define QSCALE_LOG2 (0.17677669529663687f * 1.44269504088896340736f)

typedef __attribute__((ext_vector_type(16))) _Float16 v16h;
typedef __attribute__((ext_vector_type(8)))  _Float16 v8h;
typedef __attribute__((ext_vector_type(2)))  _Float16 h2;
typedef __attribute__((ext_vector_type(8)))  float    v8f;

// --------------------------------------------------------------------------
// Kernel 1: fused QKV 1x1-conv projection.
//   q_f16  : (bm, L, 128) f16, pre-scaled by log2(e)/sqrt(hd) (WMMA B operand)
//   k_f16  : (bm, L, 128) f16                                 (WMMA A operand)
//   v_tr   : (bm, 128, L) f16, dim-major transpose            (WMMA A operand)
//   q_f32  : (bm, L, 128) f32, unscaled (for (attn+q)*0.5 fusion)
// bm = b*2 + modality. Lanes map to output channel o -> coalesced f16 writes.
// --------------------------------------------------------------------------
__global__ __launch_bounds__(256) void qkv_proj_kernel(
    const float* __restrict__ feat_ct, const float* __restrict__ feat_pet,
    const float* __restrict__ wq, const float* __restrict__ bq,
    const float* __restrict__ wk, const float* __restrict__ bk,
    const float* __restrict__ wv, const float* __restrict__ bv,
    _Float16* __restrict__ qf16, _Float16* __restrict__ kf16,
    _Float16* __restrict__ vtr,  float* __restrict__ qf32)
{
    const int bm  = blockIdx.y;
    const int b   = bm >> 1;
    const int mod = bm & 1;
    const int o   = threadIdx.x & (TD - 1);
    const int l   = blockIdx.x * 2 + (threadIdx.x >> 7);

    const float* __restrict__ fp =
        (mod ? feat_pet : feat_ct) + (size_t)b * CIN * LTOK + l;

    float aq = bq[o], ak = bk[o], av = bv[o];
    const float4* wq4 = (const float4*)(wq + o * CIN);
    const float4* wk4 = (const float4*)(wk + o * CIN);
    const float4* wv4 = (const float4*)(wv + o * CIN);

#pragma unroll
    for (int cc = 0; cc < CIN / 4; ++cc) {
        float x0 = fp[(size_t)(cc * 4 + 0) * LTOK];
        float x1 = fp[(size_t)(cc * 4 + 1) * LTOK];
        float x2 = fp[(size_t)(cc * 4 + 2) * LTOK];
        float x3 = fp[(size_t)(cc * 4 + 3) * LTOK];
        float4 a = wq4[cc]; aq += a.x * x0 + a.y * x1 + a.z * x2 + a.w * x3;
        float4 k = wk4[cc]; ak += k.x * x0 + k.y * x1 + k.z * x2 + k.w * x3;
        float4 v = wv4[cc]; av += v.x * x0 + v.y * x1 + v.z * x2 + v.w * x3;
    }

    const size_t qi = ((size_t)bm * LTOK + l) * TD + o;
    qf16[qi] = (_Float16)(aq * QSCALE_LOG2);
    kf16[qi] = (_Float16)ak;
    vtr[((size_t)bm * TD + o) * LTOK + l] = (_Float16)av;
    qf32[qi] = aq;
}

// --------------------------------------------------------------------------
// Kernel 2: flash-style cross attention, per wave: 16 query rows, one head.
// S^T = K * Q^T via v_wmma_f32_16x16x32_f16 (queries in lane columns ->
// per-query online softmax needs one shfl_xor(16)), then O^T += V^T * P^T.
// P^T B-fragment assembled from the exp'd D fragments with 4 packed-half2
// shfl_xor ops -- no LDS transpose.  Softmax runs in exp2 domain (log2e is
// folded into the f16 Q copy) so each probability is a single v_exp_f32.
// --------------------------------------------------------------------------
__global__ __launch_bounds__(256) void cross_attn_kernel(
    const _Float16* __restrict__ qf16, const _Float16* __restrict__ kf16,
    const _Float16* __restrict__ vtr,  const float* __restrict__ qf32,
    float* __restrict__ fused)
{
    const int lane = threadIdx.x & 31;
    const int wid  = threadIdx.x >> 5;      // wave 0..7
    const int n    = lane & 15;             // query column within fragment
    const int g    = lane >> 4;             // half-wave group
    const int head = blockIdx.y;
    const int bm   = blockIdx.z;
    const int kvbm = bm ^ 1;                // cross-modal: other modality, same batch
    const int q0   = blockIdx.x * 128 + wid * 16;

    // Q^T B-fragment (persistent): lane n = query col, halves i -> c = g*16 + i
    const _Float16* qp =
        qf16 + ((size_t)bm * LTOK + q0 + n) * TD + head * HD + g * 16;
    const v16h qb = *(const v16h*)qp;

    const _Float16* kb = kf16 + (size_t)kvbm * LTOK * TD + head * HD;
    const _Float16* vb = vtr + ((size_t)kvbm * TD + head * HD) * LTOK;

    float mrow = -1e30f;        // running max (log2 domain, replicated g=0/1)
    float lrow = 0.0f;          // running denom
    v8f co0 = {};               // O^T dims head*32 + (r + 8g)
    v8f co1 = {};               // O^T dims head*32 + 16 + (r + 8g)
    const v8f zf = {};

    for (int t = 0; t < LTOK / 32; ++t) {
        // ---- K A-fragments, two 16-key subtiles (row = key, halves span c)
        const _Float16* kr0 = kb + (size_t)(t * 32 + n) * TD;
        const _Float16* kr1 = kb + (size_t)(t * 32 + 16 + n) * TD;
        v8h lo0 = *(const v8h*)(kr0 + 16 * g);
        v8h hi0 = *(const v8h*)(kr0 + 16 * g + 8);
        v8h lo1 = *(const v8h*)(kr1 + 16 * g);
        v8h hi1 = *(const v8h*)(kr1 + 16 * g + 8);
        v16h ka0, ka1;
#pragma unroll
        for (int i = 0; i < 8; ++i) {
            ka0[i] = lo0[i]; ka0[i + 8] = hi0[i];
            ka1[i] = lo1[i]; ka1[i + 8] = hi1[i];
        }

        // ---- S^T tiles: st[r] = score(key = t*32 + {0,16} + r + 8g, query n)
        v8f st0 = __builtin_amdgcn_wmma_f32_16x16x32_f16(
            false, ka0, false, qb, (short)0, zf, false, false);
        v8f st1 = __builtin_amdgcn_wmma_f32_16x16x32_f16(
            false, ka1, false, qb, (short)0, zf, false, false);

        // ---- online softmax (exp2 domain) over this 32-key window
        float tmax = -1e30f;
#pragma unroll
        for (int r = 0; r < 8; ++r)
            tmax = fmaxf(tmax, fmaxf(st0[r], st1[r]));
        tmax = fmaxf(tmax, __shfl_xor(tmax, 16, 32));
        const float mnew = fmaxf(mrow, tmax);
        const float corr = __builtin_amdgcn_exp2f(mrow - mnew);
        mrow = mnew;

        float p0[8], p1[8];
        float ps = 0.0f;
#pragma unroll
        for (int r = 0; r < 8; ++r) {
            p0[r] = __builtin_amdgcn_exp2f(st0[r] - mnew);
            p1[r] = __builtin_amdgcn_exp2f(st1[r] - mnew);
            ps += p0[r] + p1[r];
        }
        ps += __shfl_xor(ps, 16, 32);
        lrow = lrow * corr + ps;
#pragma unroll
        for (int r = 0; r < 8; ++r) { co0[r] *= corr; co1[r] *= corr; }

        // ---- assemble P^T B-fragment (32 keys x 16 queries):
        // lane g=0 needs keys 0..15  = [own p0 | partner p0]
        // lane g=1 needs keys 16..31 = [partner p1 | own p1]
        // Exchange in packed f16 pairs: 4 shfl_xor(16) dword permutes.
        _Float16 h0[8], h1[8];
#pragma unroll
        for (int r = 0; r < 8; ++r) {
            h0[r] = (_Float16)p0[r];
            h1[r] = (_Float16)p1[r];
        }
        v16h pb;
#pragma unroll
        for (int j2 = 0; j2 < 4; ++j2) {
            h2 w;                                   // value partner needs
            w[0] = g ? h0[2 * j2]     : h1[2 * j2];
            w[1] = g ? h0[2 * j2 + 1] : h1[2 * j2 + 1];
            int wi = __builtin_bit_cast(int, w);
            int yi = __shfl_xor(wi, 16, 32);        // value received
            h2 y = __builtin_bit_cast(h2, yi);
            pb[2 * j2]     = g ? y[0] : h0[2 * j2];
            pb[2 * j2 + 1] = g ? y[1] : h0[2 * j2 + 1];
            pb[2 * j2 + 8] = g ? h1[2 * j2]     : y[0];
            pb[2 * j2 + 9] = g ? h1[2 * j2 + 1] : y[1];
        }

        // ---- V^T A-fragments (rows = dims, halves span the 32 keys)
        const _Float16* vr0 = vb + (size_t)n * LTOK + t * 32;
        const _Float16* vr1 = vb + (size_t)(16 + n) * LTOK + t * 32;
        v8h al0 = *(const v8h*)(vr0 + 16 * g);
        v8h ah0 = *(const v8h*)(vr0 + 16 * g + 8);
        v8h al1 = *(const v8h*)(vr1 + 16 * g);
        v8h ah1 = *(const v8h*)(vr1 + 16 * g + 8);
        v16h va0, va1;
#pragma unroll
        for (int i = 0; i < 8; ++i) {
            va0[i] = al0[i]; va0[i + 8] = ah0[i];
            va1[i] = al1[i]; va1[i + 8] = ah1[i];
        }

        // ---- O^T accumulation
        co0 = __builtin_amdgcn_wmma_f32_16x16x32_f16(
            false, va0, false, pb, (short)0, co0, false, false);
        co1 = __builtin_amdgcn_wmma_f32_16x16x32_f16(
            false, va1, false, pb, (short)0, co1, false, false);
    }

    // ---- epilogue: fused[l][d] = (attn[l][d] + q[l][d]) * 0.5  (raw L,128 order)
    const float inv = 1.0f / lrow;
    float* fr = fused + (size_t)bm * LTOK * TD;
    const float* qr = qf32 + (size_t)bm * LTOK * TD;
    const size_t row = (size_t)(q0 + n) * TD;
#pragma unroll
    for (int r = 0; r < 8; ++r) {
        const int d0 = head * HD + r + 8 * g;
        const int d1 = d0 + 16;
        fr[row + d0] = (co0[r] * inv + qr[row + d0]) * 0.5f;
        fr[row + d1] = (co1[r] * inv + qr[row + d1]) * 0.5f;
    }
}

// --------------------------------------------------------------------------
// Kernel 3: output 1x1 conv + residual. The reference "reshape" is a raw
// reinterpretation: conv-input channel d at spatial l is fused[d*4096 + l].
// wo (64x128 f32 = 32KB) staged in LDS; reads/writes coalesced over l.
// --------------------------------------------------------------------------
__global__ __launch_bounds__(256) void out_proj_kernel(
    const float* __restrict__ feat_ct, const float* __restrict__ feat_pet,
    const float* __restrict__ wo, const float* __restrict__ bo,
    const float* __restrict__ fused, float* __restrict__ out)
{
    __shared__ float swo[64 * TD];
    for (int i = threadIdx.x; i < 64 * TD; i += 256) swo[i] = wo[i];
    __syncthreads();

    const int bm   = blockIdx.y;
    const int b    = bm >> 1;
    const int mod  = bm & 1;
    const int lsub = threadIdx.x & 63;
    const int cgrp = threadIdx.x >> 6;        // 4 groups of 16 output channels
    const int l    = blockIdx.x * 64 + lsub;

    const float* fr = fused + (size_t)bm * LTOK * TD;

    float acc[16];
#pragma unroll
    for (int i = 0; i < 16; ++i) acc[i] = bo[cgrp * 16 + i];

    for (int d = 0; d < TD; ++d) {
        float x = fr[(size_t)d * LTOK + l];
#pragma unroll
        for (int i = 0; i < 16; ++i)
            acc[i] += swo[(cgrp * 16 + i) * TD + d] * x;
    }

    const float* fbase = (mod ? feat_pet : feat_ct) + (size_t)b * CIN * LTOK;
    float* obase = out + (size_t)mod * 2 * CIN * LTOK + (size_t)b * CIN * LTOK;
#pragma unroll
    for (int i = 0; i < 16; ++i) {
        const int co = cgrp * 16 + i;
        obase[(size_t)co * LTOK + l] = fbase[(size_t)co * LTOK + l] + acc[i];
    }
}

// --------------------------------------------------------------------------
extern "C" void kernel_launch(void* const* d_in, const int* in_sizes, int n_in,
                              void* d_out, int out_size, void* d_ws, size_t ws_size,
                              hipStream_t stream) {
    (void)in_sizes; (void)n_in; (void)out_size; (void)ws_size;

    const float* feat_ct  = (const float*)d_in[0];
    const float* feat_pet = (const float*)d_in[1];
    const float* wq = (const float*)d_in[2];
    const float* bq = (const float*)d_in[3];
    const float* wk = (const float*)d_in[4];
    const float* bk = (const float*)d_in[5];
    const float* wv = (const float*)d_in[6];
    const float* bv = (const float*)d_in[7];
    const float* wo = (const float*)d_in[8];
    const float* bo = (const float*)d_in[9];

    // Workspace layout (28 MB total):
    char* w = (char*)d_ws;
    const size_t nhalf = (size_t)4 * LTOK * TD;       // per f16 buffer element count
    _Float16* qf16 = (_Float16*)w;  w += nhalf * 2;   // 4 MB
    _Float16* kf16 = (_Float16*)w;  w += nhalf * 2;   // 4 MB
    _Float16* vtr  = (_Float16*)w;  w += nhalf * 2;   // 4 MB
    float*    qf32 = (float*)w;     w += nhalf * 4;   // 8 MB
    float*    fused = (float*)w;                      // 8 MB

    qkv_proj_kernel<<<dim3(LTOK / 2, 4), 256, 0, stream>>>(
        feat_ct, feat_pet, wq, bq, wk, bk, wv, bv, qf16, kf16, vtr, qf32);

    cross_attn_kernel<<<dim3(LTOK / 128, NH, 4), 256, 0, stream>>>(
        qf16, kf16, vtr, qf32, fused);

    out_proj_kernel<<<dim3(LTOK / 64, 4), 256, 0, stream>>>(
        feat_ct, feat_pet, wo, bo, fused, (float*)d_out);
}
